// Decoder_63488206569910
// MI455X (gfx1250) — compile-verified, matrix operand order checked by reading, not run on hardware
//
#include <hip/hip_runtime.h>
#include <hip/hip_bf16.h>
#include <math.h>

// ---------------------------------------------------------------------------
// CDNA5 (gfx1250, wave32) implementation.
//   * v_wmma_f32_16x16x32_bf16 for all GEMMs (bf16 in, f32 accumulate)
//   * global_load_async_to_lds_b128 to overlap the mask stream with WMMA
//   * LDS staging (padded) for the three weight slices in the fat GEMMs
// ---------------------------------------------------------------------------

typedef __attribute__((ext_vector_type(16))) __bf16 v16bf;
typedef __attribute__((ext_vector_type(8)))  float  v8f;

#define WMMA_BF16(a, b, c) \
  __builtin_amdgcn_wmma_f32_16x16x32_bf16(false, (a), false, (b), (short)0, (c), false, false)

#define BQ   512       // batch / rows of the big GEMMs
#define LQ   200       // sequence length
#define DQ   512       // embedding dim (K of all GEMMs)
#define TQ   50000     // vocabulary (N of the big GEMMs)
#define LT   13        // ceil(200/16)
#define INV_TEMP 0.04419417382415922f   // 1/sqrt(512)

// fast tanh: v_exp_f32 + v_rcp_f32, no branches (|err| ~1e-6)
static __device__ __forceinline__ float fast_tanh(float x) {
  float xc = fminf(fmaxf(x, -10.0f), 10.0f);
  float e  = __expf(2.0f * xc);
  return (e - 1.0f) * __builtin_amdgcn_rcpf(e + 1.0f);
}

static __device__ __forceinline__ unsigned pack_bf16(float a, float b) {
  union { __bf16 h[2]; unsigned u; } q;
  q.h[0] = (__bf16)a;
  q.h[1] = (__bf16)b;
  return q.u;
}

// ---------------------------------------------------------------------------
// Tile loaders: ISA 7.12.2 16-bit A-matrix 16x32 layout.
//   lane<16 : M = lane,     K packed 2/VGPR: {0..7, 16..23}
//   lane>=16: M = lane-16,  K shifted +8:    {8..15, 24..31}
// B operand mirrors this (N in lane, K packed), so the B-tile of X^T loads
// with the same code from row-major X.  Row validity is folded into the
// per-lane scale (no cndmask chain); loads are clamped in-bounds.
// ---------------------------------------------------------------------------
static __device__ __forceinline__ v16bf
load_tile_f32(const float* __restrict__ mat, int ld, int row0, int nrows,
              int k0, int lane, float scale) {
  int r = row0 + (lane & 15);
  bool valid = (r < nrows);
  int rc = valid ? r : 0;
  float s = valid ? scale : 0.0f;       // zeroes invalid rows via the multiply
  const float* p = mat + (size_t)rc * ld + k0 + ((lane < 16) ? 0 : 8);
  v16bf t;
#pragma unroll
  for (int v = 0; v < 8; ++v) {
    int k = (v < 4) ? (2 * v) : (16 + 2 * (v - 4));
    t[2 * v]     = (__bf16)(p[k] * s);
    t[2 * v + 1] = (__bf16)(p[k + 1] * s);
  }
  return t;
}

static __device__ __forceinline__ v16bf
load_tile_bf16(const __bf16* __restrict__ mat, int ld, int row0, int k0, int lane) {
  const __bf16* p = mat + (size_t)(row0 + (lane & 15)) * ld + k0 + ((lane < 16) ? 0 : 8);
  v16bf t;
#pragma unroll
  for (int v = 0; v < 8; ++v) {
    int k = (v < 4) ? (2 * v) : (16 + 2 * (v - 4));
    t[2 * v]     = p[k];
    t[2 * v + 1] = p[k + 1];
  }
  return t;
}

// ---------------------------------------------------------------------------
// K0: zero-init accumulators (norms + seq) every launch.
// ---------------------------------------------------------------------------
__global__ void k_zero(float* __restrict__ p, int n) {
  int i = blockIdx.x * blockDim.x + threadIdx.x;
  if (i < n) p[i] = 0.0f;
}

// ---------------------------------------------------------------------------
// K1: attention, fused.  One block per (M-tile i, batch b).
//  1) async-copy mask slab [nrows x 200] f32 into LDS (overlapped with WMMA)
//  2) S-tiles of (E/TEMP) E^T via bf16 WMMA over K=512 (4 accumulators/wave)
//  3) epilogue: colsum[m] += sum_l tanh(S[l,m]) * mask[l,m]  (LDS atomics)
//  4) partial GEMV: seq[b,:] += colsum_block @ E_b   (global f32 atomics)
// ---------------------------------------------------------------------------
__global__ void __launch_bounds__(128)
k_attn(const float* __restrict__ enc, const float* __restrict__ mask,
       float* __restrict__ seq) {
  __shared__ float sh[208 + 16 * LQ];   // [0..207] colsum, [208..] mask slab
  const int b    = blockIdx.y;
  const int i    = blockIdx.x;          // M tile over rows l
  const int tid  = threadIdx.x;
  const int lane = tid & 31;
  const int wave = tid >> 5;            // 0..3
  const float* E = enc + (size_t)b * LQ * DQ;
  const int l0    = 16 * i;
  const int nrows = (LQ - l0 < 16) ? (LQ - l0) : 16;

  // (1) async copy of the mask slab: nrows*200 floats, 16B chunks, per-lane
  {
    const int nchunk = (nrows * LQ) / 4;                // 50 chunks per row
    const float* gsrc = mask + ((size_t)b * LQ + l0) * LQ;
    for (int c = tid; c < nchunk; c += 128) {
      unsigned           loff  = 208u * 4u + (unsigned)c * 16u;  // LDS byte offset
      unsigned long long gaddr = (unsigned long long)(uintptr_t)gsrc
                               + (unsigned long long)c * 16ull;
      asm volatile("global_load_async_to_lds_b128 %0, %1, off"
                   :: "v"(loff), "v"(gaddr) : "memory");
    }
  }

  // zero colsum region
  for (int m = tid; m < 208; m += 128) sh[m] = 0.0f;
  __syncthreads();

  // (2) WMMA: A tile fixed (rows l0..), each wave owns j = wave + 4*jj
  v8f acc[4] = {};
  for (int k0 = 0; k0 < DQ; k0 += 32) {
    v16bf a = load_tile_f32(E, DQ, l0, LQ, k0, lane, INV_TEMP);
#pragma unroll
    for (int jj = 0; jj < 4; ++jj) {
      int j = wave + 4 * jj;            // wave-uniform -> EXEC stays full
      if (j < LT) {
        v16bf bm = load_tile_f32(E, DQ, 16 * j, LQ, k0, lane, 1.0f);
        acc[jj] = WMMA_BF16(a, bm, acc[jj]);
      }
    }
  }

  // wait for this wave's async copies, then barrier so all slabs are visible
#if __has_builtin(__builtin_amdgcn_s_wait_asynccnt)
  __builtin_amdgcn_s_wait_asynccnt(0);
#else
  asm volatile("s_wait_asynccnt 0x0" ::: "memory");
#endif
  __syncthreads();

  // (3) epilogue: tanh * mask, column partials -> LDS colsum (branch-free)
#pragma unroll
  for (int jj = 0; jj < 4; ++jj) {
    int j = wave + 4 * jj;
    if (j < LT) {
      int  mcol = 16 * j + (lane & 15);
      bool cv   = (mcol < LQ);
      int  mc   = cv ? mcol : 0;
      float part = 0.0f;
#pragma unroll
      for (int v = 0; v < 8; ++v) {
        int  lrow = v + ((lane < 16) ? 0 : 8);    // local row in slab
        bool lv   = (lrow < nrows);
        int  lr   = lv ? lrow : 0;
        float mk  = sh[208 + lr * LQ + mc];
        float w   = (lv && cv) ? mk : 0.0f;
        part += fast_tanh(acc[jj][v]) * w;
      }
      atomicAdd(&sh[mc], part);                   // ds_add_f32
    }
  }
  __syncthreads();

  // (4) partial GEMV: seq[b,d] += sum_m colsum[m] * E[m,d]
  float4 av = {0.f, 0.f, 0.f, 0.f};
  const int d0 = 4 * tid;                         // 128 threads * 4 = 512 cols
  for (int m = 0; m < LQ; ++m) {
    float c = sh[m];
    float4 e = *(const float4*)(E + (size_t)m * DQ + d0);
    av.x += c * e.x; av.y += c * e.y; av.z += c * e.z; av.w += c * e.w;
  }
  float* sp = seq + (size_t)b * DQ + d0;
  atomicAdd(sp + 0, av.x);
  atomicAdd(sp + 1, av.y);
  atomicAdd(sp + 2, av.z);
  atomicAdd(sp + 3, av.w);
}

// ---------------------------------------------------------------------------
// K3: bf16 conversion of the two 512x512 left operands (seq scaled by 1/L).
// ---------------------------------------------------------------------------
__global__ void k_cvt(const float* __restrict__ user, const float* __restrict__ seq,
                      unsigned* __restrict__ ubf2, unsigned* __restrict__ sbf2, int n4) {
  int i = blockIdx.x * blockDim.x + threadIdx.x;
  if (i < n4) {
    float4 u = ((const float4*)user)[i];
    float4 s = ((const float4*)seq)[i];
    const float sc = 1.0f / (float)LQ;
    ubf2[2 * i]     = pack_bf16(u.x, u.y);
    ubf2[2 * i + 1] = pack_bf16(u.z, u.w);
    sbf2[2 * i]     = pack_bf16(s.x * sc, s.y * sc);
    sbf2[2 * i + 1] = pack_bf16(s.z * sc, s.w * sc);
  }
}

// ---------------------------------------------------------------------------
// K4 (FINAL=false): row sums-of-squares of the 3 products (norm pass).
// K5 (FINAL=true):  recompute products, normalize, combine, tanh, store.
// One block per 16-wide T-tile; 3 weight slices staged to LDS as bf16 with
// padded stride 514 halfs (dword index = row*257 + col/2).  In the FINAL
// pass the 3x512 reciprocal norms are precomputed into LDS as well, so the
// epilogue is pure FMA + fast_tanh.
// ---------------------------------------------------------------------------
template <bool FINAL>
__global__ void __launch_bounds__(256)
k_gemm3(const float* __restrict__ emb1,   // embeddings + 512 (skip row 0)
        const float* __restrict__ Wg,
        const float* __restrict__ Wa,
        const __bf16* __restrict__ Ubf,
        const __bf16* __restrict__ Sbf,
        float* __restrict__ norms,        // [3][512]
        float* __restrict__ out) {        // [512][50000]
  __shared__ unsigned Bsh_u[3 * 16 * 257];
  __shared__ float    nsh[3 * BQ];              // inv-norms (FINAL only)
  const __bf16* Bsh = (const __bf16*)Bsh_u;     // stride 514 halfs per row
  const int t0   = blockIdx.x * 16;
  const int tid  = threadIdx.x;
  const int lane = tid & 31;
  const int wave = tid >> 5;                    // 0..7

  // stage the three 16x512 f32 slices as bf16 (float4 loads, dword LDS stores)
#pragma unroll
  for (int mat = 0; mat < 3; ++mat) {
    const float* src = (mat == 0) ? emb1 : ((mat == 1) ? Wg : Wa);
    for (int idx = tid; idx < 16 * 128; idx += 256) {
      int row = idx >> 7;
      int c4  = idx & 127;                      // float4 column
      float4 f = ((const float4*)(src + (size_t)(t0 + row) * DQ))[c4];
      unsigned* dst = &Bsh_u[(mat * 16 + row) * 257 + 2 * c4];
      dst[0] = pack_bf16(f.x, f.y);
      dst[1] = pack_bf16(f.z, f.w);
    }
  }
  if (FINAL) {
    for (int idx = tid; idx < 3 * BQ; idx += 256)
      nsh[idx] = __builtin_amdgcn_rcpf(fmaxf(sqrtf(norms[idx]), 1e-5f));
  }
  __syncthreads();

  for (int i = wave; i < BQ / 16; i += 8) {     // 32 M-tiles over 512 rows
    v8f acc1 = {}, acc2 = {}, acc3 = {};
    for (int k0 = 0; k0 < DQ; k0 += 32) {
      v16bf aU = load_tile_bf16(Ubf, DQ, 16 * i, k0, lane);
      v16bf aS = load_tile_bf16(Sbf, DQ, 16 * i, k0, lane);
      v16bf b1 = load_tile_bf16(Bsh + 0 * 16 * 514, 514, 0, k0, lane);
      v16bf b2 = load_tile_bf16(Bsh + 1 * 16 * 514, 514, 0, k0, lane);
      v16bf b3 = load_tile_bf16(Bsh + 2 * 16 * 514, 514, 0, k0, lane);
      acc1 = WMMA_BF16(aU, b1, acc1);
      acc2 = WMMA_BF16(aU, b2, acc2);
      acc3 = WMMA_BF16(aS, b3, acc3);
    }

    if (!FINAL) {
      // reduce squares across the 16 lanes of each half-wave, then one
      // atomic per row per matrix
#pragma unroll
      for (int v = 0; v < 8; ++v) {
        float s1 = acc1[v] * acc1[v];
        float s2 = acc2[v] * acc2[v];
        float s3 = acc3[v] * acc3[v];
#pragma unroll
        for (int m = 1; m < 16; m <<= 1) {
          s1 += __shfl_xor(s1, m, 32);
          s2 += __shfl_xor(s2, m, 32);
          s3 += __shfl_xor(s3, m, 32);
        }
        if ((lane & 15) == 0) {
          int r = 16 * i + v + ((lane < 16) ? 0 : 8);
          atomicAdd(&norms[0 * BQ + r], s1);
          atomicAdd(&norms[1 * BQ + r], s2);
          atomicAdd(&norms[2 * BQ + r], s3);
        }
      }
    } else {
      int t = t0 + (lane & 15);                 // < 50000 (3125*16 == 50000)
#pragma unroll
      for (int v = 0; v < 8; ++v) {
        int r = 16 * i + v + ((lane < 16) ? 0 : 8);
        float val = acc1[v] * nsh[0 * BQ + r]
                  + acc2[v] * nsh[1 * BQ + r]
                  + acc3[v] * (0.5f * nsh[2 * BQ + r]);
        out[(size_t)r * TQ + t] = fast_tanh(val);
      }
    }
  }
}

// ---------------------------------------------------------------------------
// Launch. Workspace: norms [3*512] | seq [512*512] | Ubf bf16[512*512] |
// Sbf bf16[512*512]  (~2.1 MB). norms+seq zeroed every call.
// ---------------------------------------------------------------------------
extern "C" void kernel_launch(void* const* d_in, const int* in_sizes, int n_in,
                              void* d_out, int out_size, void* d_ws, size_t ws_size,
                              hipStream_t stream) {
  const float* user = (const float*)d_in[0];   // [512,512]
  const float* emb  = (const float*)d_in[1];   // [50001,512]
  const float* enc  = (const float*)d_in[2];   // [512,200,512]
  const float* mask = (const float*)d_in[3];   // [512,200,200]
  const float* Wg   = (const float*)d_in[4];   // [50000,512]
  const float* Wa   = (const float*)d_in[5];   // [50000,512]
  float* out = (float*)d_out;                  // [512,50000]

  float*  norms = (float*)d_ws;
  float*  seq   = norms + 3 * BQ;
  __bf16* Ubf   = (__bf16*)(seq + BQ * DQ);
  __bf16* Sbf   = Ubf + BQ * DQ;

  int nz = 3 * BQ + BQ * DQ;                   // norms + seq, contiguous
  k_zero<<<(nz + 255) / 256, 256, 0, stream>>>(norms, nz);

  k_attn<<<dim3(LT, BQ), 128, 0, stream>>>(enc, mask, seq);
  k_cvt<<<(BQ * DQ / 4 + 255) / 256, 256, 0, stream>>>(
      user, seq, (unsigned*)Ubf, (unsigned*)Sbf, BQ * DQ / 4);

  k_gemm3<false><<<TQ / 16, 256, 0, stream>>>(emb + DQ, Wg, Wa, Ubf, Sbf, norms, out);
  k_gemm3<true ><<<TQ / 16, 256, 0, stream>>>(emb + DQ, Wg, Wa, Ubf, Sbf, norms, out);
}
